// AttentionBlock_70291434766465
// MI455X (gfx1250) — compile-verified, matrix operand order checked by reading, not run on hardware
//
#include <hip/hip_runtime.h>

#define NN    128
#define EE    64
#define HID   256
#define NH    8
#define DH    32
#define R16   16   // L*L
#define RSQRT2      0.70710678118654752f
#define INV_SQRT_DH 0.17677669529663687f

typedef __attribute__((ext_vector_type(16))) __bf16 v16bf;
typedef __attribute__((ext_vector_type(8)))  __bf16 v8bf;
typedef __attribute__((ext_vector_type(4)))  __bf16 v4bf;
typedef __attribute__((ext_vector_type(8)))  float  v8f;

// --------- CDNA5 async global->LDS copy (ASYNCcnt path), with fallback ------
#if defined(__gfx1250__) && __has_builtin(__builtin_amdgcn_global_load_async_to_lds_b128)
#define HAVE_ASYNC_LDS 1
#else
#define HAVE_ASYNC_LDS 0
#endif

#if HAVE_ASYNC_LDS
typedef int i4v __attribute__((vector_size(16)));
typedef __attribute__((address_space(1))) i4v* gptr128;
typedef __attribute__((address_space(3))) i4v* lptr128;

__device__ __forceinline__ void g2l_b128(const __bf16* g, __bf16* l) {
  __builtin_amdgcn_global_load_async_to_lds_b128((gptr128)g, (lptr128)l, 0, 0);
}
__device__ __forceinline__ void g2l_wait() {
#if __has_builtin(__builtin_amdgcn_s_wait_asynccnt)
  __builtin_amdgcn_s_wait_asynccnt(0);
#else
  asm volatile("s_wait_asynccnt 0" ::: "memory");
#endif
}
#else
__device__ __forceinline__ void g2l_b128(const __bf16* g, __bf16* l) {
  *(v8bf*)l = *(const v8bf*)g;
}
__device__ __forceinline__ void g2l_wait() {}
#endif

// ---------------- WMMA fragment helpers (wave32, 16x16x32 bf16) -------------
__device__ __forceinline__ int fragK(int i, int half) {
  return ((i & 8) << 1) | (half << 3) | (i & 7);
}

__device__ __forceinline__ v16bf load_a_lds(const __bf16* base, int stride, int lane) {
  const int half = lane >> 4, m = lane & 15;
  v16bf a;
#pragma unroll
  for (int i = 0; i < 16; ++i) a[i] = base[m * stride + fragK(i, half)];
  return a;
}

__device__ __forceinline__ v16bf load_b_lds(const __bf16* base, int stride, int lane) {
  const int half = lane >> 4, nn = lane & 15;
  v16bf b;
#pragma unroll
  for (int i = 0; i < 16; ++i) b[i] = base[fragK(i, half) * stride + nn];
  return b;
}

__device__ __forceinline__ v16bf load_b_wrow(const float* __restrict__ w, int o0, int k0, int lane) {
  const int half = lane >> 4, nn = lane & 15;
  const float* row = w + (size_t)(o0 + nn) * HID + k0;
  v16bf b;
#pragma unroll
  for (int i = 0; i < 16; ++i) b[i] = (__bf16)row[fragK(i, half)];
  return b;
}

__device__ __forceinline__ v8f wmma_bf16(v16bf a, v16bf b, v8f c) {
  return __builtin_amdgcn_wmma_f32_16x16x32_bf16(false, a, false, b, (short)0, c, false, false);
}

// ---------------- 4x4 matrix exponential (scaling & squaring + Horner) ------
__device__ __forceinline__ void mm4(const float* X, const float* Y, float* Z) {
#pragma unroll
  for (int r = 0; r < 4; ++r)
#pragma unroll
    for (int c = 0; c < 4; ++c) {
      float acc = 0.f;
#pragma unroll
      for (int k = 0; k < 4; ++k) acc = fmaf(X[r * 4 + k], Y[k * 4 + c], acc);
      Z[r * 4 + c] = acc;
    }
}

__device__ void expm4(const float* A, float* E) {
  float nrm = 0.f;
#pragma unroll
  for (int r = 0; r < 4; ++r) {
    float s = 0.f;
#pragma unroll
    for (int c = 0; c < 4; ++c) s += fabsf(A[r * 4 + c]);
    nrm = fmaxf(nrm, s);
  }
  int s = 0;
  float sc = 1.f;
  while (nrm * sc > 0.5f && s < 30) { sc *= 0.5f; ++s; }
  float B[16], M[16], T[16];
#pragma unroll
  for (int i = 0; i < 16; ++i) B[i] = A[i] * sc;
#pragma unroll
  for (int i = 0; i < 16; ++i) M[i] = (i % 5 == 0) ? 1.f : 0.f;
#pragma unroll
  for (int k = 8; k >= 1; --k) {          // M = I + (B*M)/k
    mm4(B, M, T);
    const float inv = 1.f / (float)k;
#pragma unroll
    for (int i = 0; i < 16; ++i) M[i] = ((i % 5 == 0) ? 1.f : 0.f) + T[i] * inv;
  }
  for (int q = 0; q < s; ++q) {
    mm4(M, M, T);
#pragma unroll
    for (int i = 0; i < 16; ++i) M[i] = T[i];
  }
#pragma unroll
  for (int i = 0; i < 16; ++i) E[i] = M[i];
}

// ---------------- Kernel 1: per-(n,e) trace std --> inv_std -----------------
__global__ void __launch_bounds__(256) stats_kernel(const float* __restrict__ in,
                                                    float* __restrict__ inv_std) {
  const int ne = blockIdx.x;
  const int c = threadIdx.x;
  const float* p = in + ((size_t)ne * HID + c) * R16;
  const float t = p[0] + p[5] + p[10] + p[15];
  __shared__ float s1[256], s2[256];
  s1[c] = t; s2[c] = t * t;
  __syncthreads();
  for (int off = 128; off > 0; off >>= 1) {
    if (c < off) { s1[c] += s1[c + off]; s2[c] += s2[c + off]; }
    __syncthreads();
  }
  if (c == 0) {
    const float sum = s1[0], sumsq = s2[0];
    const float mean = sum * (1.f / 256.f);
    float var = (sumsq - sum * mean) * (1.f / 255.f);
    var = fmaxf(var, 0.f);
    inv_std[ne] = 1.f / (sqrtf(var) + 1e-8f);
  }
}

// ---------------- Kernel 2: QKV projection (WMMA) ---------------------------
// Global layouts chosen so kernel 3 can do straight row copies into LDS:
//   q: per (n,h)  [(e*4+a)][ (d*4+b) ]   (256 x 128)
//   k: per (n,h)  [(d*4+b)][ (f*4+c) ]   (128 x 256), f = this block's e
//   v: per (n,h)  [(f*4+b)][ (d*4+c) ]   (256 x 128), f = this block's e
__global__ void __launch_bounds__(256) qkv_kernel(
    const float* __restrict__ in, const float* __restrict__ rms,
    const float* __restrict__ inv_std, const float* __restrict__ w_in,
    __bf16* __restrict__ qb, __bf16* __restrict__ kb, __bf16* __restrict__ vb) {
  __shared__ __bf16 xs[16][264];
  const int ne = blockIdx.x;
  const int n = ne / EE, e = ne % EE;
  const int tid = threadIdx.x;
  const int lane = tid & 31, w = tid >> 5;
  {
    const int c = tid;
    const float sc = rms[c] * inv_std[ne];
    const float* src = in + ((size_t)ne * HID + c) * R16;
#pragma unroll
    for (int r = 0; r < 16; ++r) xs[r][c] = (__bf16)(src[r] * sc);
  }
  __syncthreads();
  const int half = lane >> 4, nn2 = lane & 15;
  for (int tt = 0; tt < 6; ++tt) {              // 48 output tiles / 8 waves
    const int o0 = (w * 6 + tt) * 16;
    __builtin_prefetch(&w_in[(size_t)o0 * HID], 0, 1);
    v8f acc = {0.f, 0.f, 0.f, 0.f, 0.f, 0.f, 0.f, 0.f};
#pragma unroll
    for (int kc = 0; kc < 8; ++kc) {
      v16bf A = load_a_lds(&xs[0][kc * 32], 264, lane);
      v16bf B = load_b_wrow(w_in, o0, kc * 32, lane);
      acc = wmma_bf16(A, B, acc);
    }
    const int o = o0 + nn2;
    const int part = o >> 8, oi = o & 255, h = oi >> 5, d = oi & 31;
    const size_t nhb = (size_t)n * NH + h;
    v4bf lo, hi;
#pragma unroll
    for (int i = 0; i < 4; ++i) { lo[i] = (__bf16)acc[i]; hi[i] = (__bf16)acc[i + 4]; }
    if (part == 0) {             // q rows (e*4+a), a = 2*half, 2*half+1
      __bf16* base = qb + (nhb * 256 + e * 4 + 2 * half) * 128 + d * 4;
      *(v4bf*)base = lo; *(v4bf*)(base + 128) = hi;
    } else if (part == 1) {      // k rows (d*4+b), b = 2*half, 2*half+1; cols (f*4+c)
      __bf16* base = kb + (nhb * 128 + d * 4 + 2 * half) * 256 + e * 4;
      *(v4bf*)base = lo; *(v4bf*)(base + 256) = hi;
    } else {                     // v rows (f*4+b), b = 2*half, 2*half+1; cols (d*4+c)
      __bf16* base = vb + (nhb * 256 + e * 4 + 2 * half) * 128 + d * 4;
      *(v4bf*)base = lo; *(v4bf*)(base + 128) = hi;
    }
  }
}

// ---------------- Kernel 3: attention per (n,h) -----------------------------
__global__ void __launch_bounds__(512) attn_kernel(
    const __bf16* __restrict__ qb, const __bf16* __restrict__ kb,
    const __bf16* __restrict__ vb, const float* __restrict__ radial,
    __bf16* __restrict__ attout) {
  extern __shared__ char smem[];
  __bf16* Qs  = (__bf16*)smem;                  // [256][136]
  __bf16* Ks  = Qs + 256 * 136;                 // [128][264]
  __bf16* Vs  = Ks + 128 * 264;                 // [256][136]
  float*  wqk = (float*)(Vs + 256 * 136);       // [16 waves][16][33]
  __bf16* atts = (__bf16*)(wqk + 16 * 16 * 33); // [16 waves][16][40]
  float*  denom = (float*)(atts + 16 * 16 * 40);// [64]

  const int tid = threadIdx.x;
  const int lane = tid & 31;
  const int w = tid >> 5;
  const int nh = blockIdx.x;
  const int n = nh / NH;
  const int h = nh - n * NH;
  const size_t base = (size_t)nh * (256 * 128);
  const __bf16* qsrc = qb + base;
  const __bf16* ksrc = kb + (size_t)nh * (128 * 256);
  const __bf16* vsrc = vb + base;

  // Row-wise async copies into padded LDS (16B chunks, ASYNCcnt tracked)
  for (int idx = tid; idx < 4096; idx += 512) { // Q: 256 rows x 128
    const int row = idx >> 4, c8 = (idx & 15) * 8;
    g2l_b128(qsrc + row * 128 + c8, Qs + row * 136 + c8);
  }
  for (int idx = tid; idx < 4096; idx += 512) { // K: 128 rows x 256
    const int row = idx >> 5, c8 = (idx & 31) * 8;
    g2l_b128(ksrc + row * 256 + c8, Ks + row * 264 + c8);
  }
  for (int idx = tid; idx < 4096; idx += 512) { // V: 256 rows x 128
    const int row = idx >> 4, c8 = (idx & 15) * 8;
    g2l_b128(vsrc + row * 128 + c8, Vs + row * 136 + c8);
  }
  if (tid < 64) denom[tid] = 0.f;
  g2l_wait();
  __syncthreads();

  const int half = lane >> 4, m = lane & 15;

  v16bf Aq[4];                                  // loop-invariant Q fragments
#pragma unroll
  for (int kc = 0; kc < 4; ++kc)
    Aq[kc] = load_a_lds(&Qs[(w * 16) * 136 + kc * 32], 136, lane);

  v8f outacc[8];
#pragma unroll
  for (int t = 0; t < 8; ++t) outacc[t] = (v8f){0.f, 0.f, 0.f, 0.f, 0.f, 0.f, 0.f, 0.f};

  for (int f0 = 0; f0 < 8; ++f0) {              // 8 f's per chunk
#pragma unroll
    for (int ct = 0; ct < 2; ++ct) {
      v8f acc = {0.f, 0.f, 0.f, 0.f, 0.f, 0.f, 0.f, 0.f};
#pragma unroll
      for (int kc = 0; kc < 4; ++kc) {
        v16bf B = load_b_lds(&Ks[(kc * 32) * 264 + f0 * 32 + ct * 16], 264, lane);
        acc = wmma_bf16(Aq[kc], B, acc);
      }
      float* wp = &wqk[w * 528];
#pragma unroll
      for (int i = 0; i < 8; ++i) wp[(i + 8 * half) * 33 + ct * 16 + m] = acc[i];
    }
    __syncthreads();
    {                                           // expm: one 4x4 per lane
      const int e_loc = lane >> 3, fl = lane & 7;
      const float* wp = &wqk[w * 528];
      float A4[16], E4[16];
#pragma unroll
      for (int r = 0; r < 4; ++r)
#pragma unroll
        for (int c = 0; c < 4; ++c)
          A4[r * 4 + c] = wp[(e_loc * 4 + r) * 33 + fl * 4 + c] * INV_SQRT_DH;
      expm4(A4, E4);
      const float tr = E4[0] + E4[5] + E4[10] + E4[15];
      atomicAdd(&denom[w * 4 + e_loc], tr);     // ds_add_f32
      const float mk = radial[n * EE + f0 * 8 + fl];
      __bf16* ap = &atts[w * 640];
#pragma unroll
      for (int r = 0; r < 4; ++r)
#pragma unroll
        for (int c = 0; c < 4; ++c)
          ap[(e_loc * 4 + r) * 40 + fl * 4 + c] = (__bf16)(E4[r * 4 + c] * mk);
    }
    __syncthreads();
    v16bf Aatt = load_a_lds(&atts[w * 640], 40, lane);
#pragma unroll
    for (int t = 0; t < 8; ++t) {
      v16bf B = load_b_lds(&Vs[(f0 * 32) * 136 + t * 16], 136, lane);
      outacc[t] = wmma_bf16(Aatt, B, outacc[t]);
    }
  }
  __syncthreads();

  float fct8[8];
#pragma unroll
  for (int i = 0; i < 8; ++i) {
    const int row = i + 8 * half;
    fct8[i] = 1.f / (denom[w * 4 + (row >> 2)] + 1e-6f);
  }
#pragma unroll
  for (int t = 0; t < 8; ++t) {
    const int col = t * 16 + m;                 // d*4+c
    const int d = col >> 2, c = col & 3;
#pragma unroll
    for (int i = 0; i < 8; ++i) {
      const int row = i + 8 * half;
      const int e = w * 4 + (row >> 2);
      const int a = row & 3;
      const size_t o = (((size_t)n * EE + e) * R16 + a * 4 + c) * HID + h * DH + d;
      attout[o] = (__bf16)(outacc[t][i] * fct8[i]);
    }
  }
}

// ---------------- Kernel 4: output projection + residual --------------------
__global__ void __launch_bounds__(256) proj_kernel(
    const __bf16* __restrict__ ao, const float* __restrict__ w_out,
    const float* __restrict__ in, float* __restrict__ out) {
  __shared__ __bf16 xs[16][264];
  const int ne = blockIdx.x;
  const int tid = threadIdx.x, lane = tid & 31, w = tid >> 5;
  for (int idx = tid; idx < 512; idx += 256) {  // 16 rows x 256, 16B chunks
    const int r = idx >> 5, c8 = (idx & 31) * 8;
    g2l_b128(ao + ((size_t)ne * R16 + r) * HID + c8, &xs[r][c8]);
  }
  g2l_wait();
  __syncthreads();
  const int half = lane >> 4, nn2 = lane & 15;
#pragma unroll
  for (int tt = 0; tt < 2; ++tt) {
    const int o0 = (w * 2 + tt) * 16;
    __builtin_prefetch(&w_out[(size_t)o0 * HID], 0, 1);
    v8f acc = {0.f, 0.f, 0.f, 0.f, 0.f, 0.f, 0.f, 0.f};
#pragma unroll
    for (int kc = 0; kc < 8; ++kc) {
      v16bf A = load_a_lds(&xs[0][kc * 32], 264, lane);
      v16bf B = load_b_wrow(w_out, o0, kc * 32, lane);
      acc = wmma_bf16(A, B, acc);
    }
    const int o = o0 + nn2;
    const size_t b = ((size_t)ne * HID + o) * R16 + half * 8;
    float4 lo, hi;
    lo.x = (acc[0] + in[b + 0]) * RSQRT2; lo.y = (acc[1] + in[b + 1]) * RSQRT2;
    lo.z = (acc[2] + in[b + 2]) * RSQRT2; lo.w = (acc[3] + in[b + 3]) * RSQRT2;
    hi.x = (acc[4] + in[b + 4]) * RSQRT2; hi.y = (acc[5] + in[b + 5]) * RSQRT2;
    hi.z = (acc[6] + in[b + 6]) * RSQRT2; hi.w = (acc[7] + in[b + 7]) * RSQRT2;
    *(float4*)(out + b) = lo;
    *(float4*)(out + b + 4) = hi;
  }
}

// ---------------- Launch ----------------------------------------------------
#define SMEM_ATTN ((256*136 + 128*264 + 256*136) * 2 + (16*16*33) * 4 + (16*16*40) * 2 + 64 * 4)

extern "C" void kernel_launch(void* const* d_in, const int* in_sizes, int n_in,
                              void* d_out, int out_size, void* d_ws, size_t ws_size,
                              hipStream_t stream) {
  (void)in_sizes; (void)n_in; (void)out_size; (void)ws_size;
  const float* in     = (const float*)d_in[0];
  const float* radial = (const float*)d_in[1];
  const float* w_in   = (const float*)d_in[3];
  const float* w_out  = (const float*)d_in[4];
  const float* rms    = (const float*)d_in[5];
  float* out = (float*)d_out;

  char* ws = (char*)d_ws;
  const size_t QKV_ELEMS = (size_t)NN * NH * 256 * 128;     // 33,554,432 per tensor
  float*  inv_std = (float*)ws;
  __bf16* qb = (__bf16*)(ws + (64 << 10));
  __bf16* kb = qb + QKV_ELEMS;
  __bf16* vb = kb + QKV_ELEMS;
  __bf16* ao = vb + QKV_ELEMS;                              // (n,e,16,HID)

  (void)hipFuncSetAttribute(reinterpret_cast<const void*>(attn_kernel),
                            hipFuncAttributeMaxDynamicSharedMemorySize, SMEM_ATTN);

  stats_kernel<<<NN * EE, 256, 0, stream>>>(in, inv_std);
  qkv_kernel<<<NN * EE, 256, 0, stream>>>(in, rms, inv_std, w_in, qb, kb, vb);
  attn_kernel<<<NN * NH, 512, SMEM_ATTN, stream>>>(qb, kb, vb, radial, ao);
  proj_kernel<<<NN * EE, 256, 0, stream>>>(ao, w_out, in, out);
}
